// LSTMDecoder_19799799235160
// MI455X (gfx1250) — matrix operand order repaired and checked by reading, not compiled, not run on hardware
//
#include <hip/hip_runtime.h>
#include <hip/hip_bf16.h>

// Problem sizes (fixed by the reference)
#define BSZ     256
#define TSTEPS  512
#define HDIM    256
#define LDIM    128
#define HPAD    264        // padded LDS row stride (bf16 elems): dword stride 132 ≡ 4 (mod 64) -> conflict-free A loads

typedef __attribute__((ext_vector_type(16))) __bf16 v16bf;
typedef __attribute__((ext_vector_type(8)))  float  v8f;

union Frag16 { v16bf v; uint4 q[2]; unsigned short us[16]; };

__device__ __forceinline__ unsigned short f2bf(float f) {
  unsigned int u = __float_as_uint(f);
  u += 0x7FFFu + ((u >> 16) & 1u);          // round-to-nearest-even
  return (unsigned short)(u >> 16);
}
__device__ __forceinline__ float fsig(float x) {
  float e = __builtin_amdgcn_exp2f(-1.4426950408889634f * x);
  return __builtin_amdgcn_rcpf(1.0f + e);
}
__device__ __forceinline__ float ftanh(float x) {
  x = fminf(fmaxf(x, -10.0f), 10.0f);
  float e = __builtin_amdgcn_exp2f(-2.8853900817779268f * x);   // exp(-2x)
  return (1.0f - e) * __builtin_amdgcn_rcpf(1.0f + e);
}

// B-fragment load, byte addressing: pkc is the kernarg base (global
// provenance -> global_load_b128), voffB is the divergent per-lane byte
// offset (+ laundered zero), cByte is a compile-time byte constant that
// folds into the instruction's signed 24-bit offset field.
__device__ __forceinline__ void load_frag(Frag16& f,
                                          const char* __restrict__ pkc,
                                          unsigned int voffB, int cByte) {
  const uint4* p = (const uint4*)(pkc + voffB + cByte);
  f.q[0] = p[0];
  f.q[1] = p[1];
}

// ---------------------------------------------------------------------------
// Pack W (row-major [1024 x 256] f32, z = h @ W^T) into bf16 WMMA-B-fragment
// order: per (ntile, ktile, lane) the 16 bf16 a lane feeds to
// v_wmma_f32_16x16x32_bf16 are contiguous (32 B).  B-layout used here: N = lane%16,
// K = r + (lane>=16 ? 16 : 0) within the 32-wide K tile.
// ---------------------------------------------------------------------------
__global__ void pack_weights(const float* __restrict__ Whh0,
                             const float* __restrict__ Wih1,
                             const float* __restrict__ Whh1,
                             unsigned short* __restrict__ pk) {
  int tid = blockIdx.x * blockDim.x + threadIdx.x;   // 3 * 64 * 8 * 32 = 49152
  int mat = tid >> 14;
  int rem = tid & 16383;
  int nt  = rem >> 8;            // 0..63
  int kt  = (rem >> 5) & 7;      // 0..7
  int L   = rem & 31;            // lane
  const float* W = (mat == 0) ? Whh0 : (mat == 1) ? Wih1 : Whh1;
  int n  = nt * 16 + (L & 15);
  int k0 = kt * 32 + ((L >= 16) ? 16 : 0);
  const float4* src = (const float4*)(W + (size_t)n * HDIM + k0);
  unsigned short us[16];
#pragma unroll
  for (int v = 0; v < 4; ++v) {
    float4 f = src[v];
    us[v*4+0] = f2bf(f.x); us[v*4+1] = f2bf(f.y);
    us[v*4+2] = f2bf(f.z); us[v*4+3] = f2bf(f.w);
  }
  unsigned int w8[8];
#pragma unroll
  for (int i = 0; i < 8; ++i)
    w8[i] = (unsigned int)us[2*i] | ((unsigned int)us[2*i+1] << 16);
  uint4* dst = (uint4*)(pk + (size_t)mat * 262144 + ((size_t)(nt*8 + kt)*32 + L) * 16);
  dst[0] = make_uint4(w8[0], w8[1], w8[2], w8[3]);
  dst[1] = make_uint4(w8[4], w8[5], w8[6], w8[7]);
}

// ---------------------------------------------------------------------------
// Persistent LSTM: one workgroup (16 wave32 = 512 threads) per 16-row batch
// tile; the whole T=512 chain runs in-kernel with only intra-WGP barriers.
// Wave w owns h-columns [16w, 16w+16) and the matching i/f/g/o gate tiles,
// so the gate nonlinearity is lane/register-local.  4 waves per SIMD hide
// the L2 latency of the weight streams.
// ---------------------------------------------------------------------------
__global__ void __launch_bounds__(512)
lstm_persistent(const float* __restrict__ latent, const float* __restrict__ W_lh,
                const float* __restrict__ b_lh,
                const float* __restrict__ b_ih0, const float* __restrict__ b_hh0,
                const float* __restrict__ b_ih1, const float* __restrict__ b_hh1,
                const float* __restrict__ W_ho,  const float* __restrict__ b_ho,
                const unsigned short* __restrict__ pk,
                float* __restrict__ out) {
  __shared__ unsigned short sh0[2][16 * HPAD];   // layer-0 h (bf16), ping-pong
  __shared__ unsigned short sh1[2][16 * HPAD];   // layer-1 h (bf16), ping-pong
  __shared__ float shOut[16][16];                // per-wave output partials

  const int tid  = threadIdx.x;
  const int wv   = tid >> 5;        // wave 0..15
  const int lane = tid & 31;
  const int m    = lane & 15;       // N-column / M-row within tile
  const int hi   = lane >> 4;       // lane half
  const int wg   = blockIdx.x;      // batch tile 0..15
  const char* pkc = (const char*)pk;

  // ---- h_init = latent @ W_lh^T + b_lh (tiny: 16x256, K=128), into both buffers
  for (int e = tid; e < 16 * HDIM; e += 512) {
    int row = e >> 8, col = e & 255;
    const float4* lr = (const float4*)(latent + (size_t)(wg*16 + row) * LDIM);
    const float4* wr = (const float4*)(W_lh + (size_t)col * LDIM);
    float acc = b_lh[col];
#pragma unroll 8
    for (int k = 0; k < LDIM/4; ++k) {
      float4 a = lr[k], b = wr[k];
      acc += a.x*b.x + a.y*b.y + a.z*b.z + a.w*b.w;
    }
    unsigned short us = f2bf(acc);
    sh0[0][row*HPAD + col] = us;
    sh1[0][row*HPAD + col] = us;
  }

  // ---- per-lane constants: fused biases for this wave's gate columns
  float bias0[4], bias1[4];
#pragma unroll
  for (int g = 0; g < 4; ++g) {
    int col = g*256 + wv*16 + m;
    bias0[g] = b_ih0[col] + b_hh0[col];
    bias1[g] = b_ih1[col] + b_hh1[col];
  }
  const float whoV = W_ho[wv*16 + m];
  const float bho  = b_ho[0];

  float c0s[8], c1s[8];
#pragma unroll
  for (int i = 0; i < 8; ++i) { c0s[i] = 0.0f; c1s[i] = 0.0f; }

  // per-lane base BYTE offset into the packed-weight arena; the per-fragment
  // part (mat*524288 + g*131072 + kt*1024 bytes) is a compile-time constant
  // folded into each load's 24-bit instruction offset.
  const unsigned int voffBase = (unsigned int)(wv*8192 + lane*32);
  unsigned int vzero = 0;    // laundered each step to keep loads in the loop

  __syncthreads();

  int p = 0;
  for (int t = 0; t < TSTEPS; ++t) {
    const int q = p ^ 1;

    asm volatile("" : "+v"(vzero));            // opaque 0: defeats LICM/spill
    const unsigned int voff = voffBase + vzero;

    // ========== layer 0: z0 = h0 @ W_hh0^T + b0 ==========
    v8f acc[4];
#pragma unroll
    for (int g = 0; g < 4; ++g) acc[g] = (v8f)bias0[g];   // bias folded into C
#pragma unroll
    for (int kt = 0; kt < 8; ++kt) {
      Frag16 A;                                  // A-layout (matches packed B): K {0..15}/{16..31}
      int abase = m*HPAD + kt*32 + hi*16;
      A.q[0] = *(const uint4*)&sh0[p][abase];
      A.q[1] = *(const uint4*)&sh0[p][abase + 8];
      Frag16 Bf[4];
#pragma unroll
      for (int g = 0; g < 4; ++g)
        load_frag(Bf[g], pkc, voff, g*131072 + kt*1024);
#pragma unroll
      for (int g = 0; g < 4; ++g)
        acc[g] = __builtin_amdgcn_wmma_f32_16x16x32_bf16(
            false, A.v, false, Bf[g].v, (short)0, acc[g], false, false);
    }
    // gate math layer 0 -> h0_new (bf16) into sh0[q]
#pragma unroll
    for (int r = 0; r < 8; ++r) {
      float c = fsig(acc[1][r])*c0s[r] + fsig(acc[0][r])*ftanh(acc[2][r]);
      c0s[r] = c;
      float h = fsig(acc[3][r])*ftanh(c);
      sh0[q][(r + hi*8)*HPAD + wv*16 + m] = f2bf(h);
    }
    __syncthreads();

    // ========== layer 1: z1 = h0_new @ W_ih1^T + h1 @ W_hh1^T + b1 ==========
#pragma unroll
    for (int g = 0; g < 4; ++g) acc[g] = (v8f)bias1[g];
#pragma unroll
    for (int kt = 0; kt < 8; ++kt) {
      Frag16 A0, A1;
      int abase = m*HPAD + kt*32 + hi*16;
      A0.q[0] = *(const uint4*)&sh0[q][abase];
      A0.q[1] = *(const uint4*)&sh0[q][abase + 8];
      A1.q[0] = *(const uint4*)&sh1[p][abase];
      A1.q[1] = *(const uint4*)&sh1[p][abase + 8];
      Frag16 B0[4], B1[4];
#pragma unroll
      for (int g = 0; g < 4; ++g) {
        load_frag(B0[g], pkc, voff,  524288 + g*131072 + kt*1024);
        load_frag(B1[g], pkc, voff, 1048576 + g*131072 + kt*1024);
      }
#pragma unroll
      for (int g = 0; g < 4; ++g) {
        acc[g] = __builtin_amdgcn_wmma_f32_16x16x32_bf16(
            false, A0.v, false, B0[g].v, (short)0, acc[g], false, false);
        acc[g] = __builtin_amdgcn_wmma_f32_16x16x32_bf16(
            false, A1.v, false, B1[g].v, (short)0, acc[g], false, false);
      }
    }
    // gate math layer 1 -> h1_new into sh1[q]; fused O=1 output projection
    float psum[8];
#pragma unroll
    for (int r = 0; r < 8; ++r) {
      float c = fsig(acc[1][r])*c1s[r] + fsig(acc[0][r])*ftanh(acc[2][r]);
      c1s[r] = c;
      float h = fsig(acc[3][r])*ftanh(c);
      sh1[q][(r + hi*8)*HPAD + wv*16 + m] = f2bf(h);
      psum[r] = h * whoV;
    }
    // reduce over the 16 lanes of each half (rows r+8*hi)
#pragma unroll
    for (int r = 0; r < 8; ++r) {
      float s = psum[r];
      s += __shfl_xor(s, 1, 32);
      s += __shfl_xor(s, 2, 32);
      s += __shfl_xor(s, 4, 32);
      s += __shfl_xor(s, 8, 32);
      psum[r] = s;
    }
    if (m == 0) {
#pragma unroll
      for (int r = 0; r < 8; ++r) shOut[wv][hi*8 + r] = psum[r];
    }
    __syncthreads();

    if (wv == 0 && lane < 16) {
      float s = bho;
#pragma unroll
      for (int w2 = 0; w2 < 16; ++w2) s += shOut[w2][lane];
      out[(size_t)(wg*16 + lane) * TSTEPS + t] = s;   // [B, T, 1]
    }
    p = q;
  }
}

extern "C" void kernel_launch(void* const* d_in, const int* in_sizes, int n_in,
                              void* d_out, int out_size, void* d_ws, size_t ws_size,
                              hipStream_t stream) {
  (void)in_sizes; (void)n_in; (void)out_size; (void)ws_size;
  const float* latent = (const float*)d_in[0];
  const float* W_lh   = (const float*)d_in[1];
  const float* b_lh   = (const float*)d_in[2];
  // d_in[3] = W_ih0 unused (layer-0 input is all-zero)
  const float* W_hh0  = (const float*)d_in[4];
  const float* b_ih0  = (const float*)d_in[5];
  const float* b_hh0  = (const float*)d_in[6];
  const float* W_ih1  = (const float*)d_in[7];
  const float* W_hh1  = (const float*)d_in[8];
  const float* b_ih1  = (const float*)d_in[9];
  const float* b_hh1  = (const float*)d_in[10];
  const float* W_ho   = (const float*)d_in[11];
  const float* b_ho   = (const float*)d_in[12];
  unsigned short* pk  = (unsigned short*)d_ws;   // 3 * 1024*256 bf16 = 1.5 MB

  pack_weights<<<192, 256, 0, stream>>>(W_hh0, W_ih1, W_hh1, pk);
  lstm_persistent<<<16, 512, 0, stream>>>(latent, W_lh, b_lh, b_ih0, b_hh0,
                                          b_ih1, b_hh1, W_ho, b_ho, pk,
                                          (float*)d_out);
}
